// ViT_3822520894102
// MI455X (gfx1250) — matrix operand order repaired and checked by reading, not compiled, not run on hardware
//
#include <hip/hip_runtime.h>

typedef unsigned short u16;
typedef __attribute__((ext_vector_type(16))) __bf16 v16bf;
typedef __attribute__((ext_vector_type(8)))  float  v8f;

// ---------------- helpers ----------------

__device__ __forceinline__ u16 f2bf(float f) {
    union { float f; unsigned int u; } v; v.f = f;
    unsigned int r = v.u + 0x7FFFu + ((v.u >> 16) & 1u);   // round-to-nearest-even
    return (u16)(r >> 16);
}

__device__ __forceinline__ uint4 zero4() {
    uint4 z; z.x = 0u; z.y = 0u; z.z = 0u; z.w = 0u; return z;
}

// CDNA5 async global->LDS copy (16B per lane), tracked by ASYNCcnt.
// vdst = VGPR holding LDS byte address (low 32 bits of generic LDS pointer),
// vaddr = 64-bit global address. EXEC-predicated per lane.
__device__ __forceinline__ void async_copy_b128(const u16* lds_ptr, const u16* gptr) {
    unsigned loff = (unsigned)(unsigned long long)lds_ptr;
    asm volatile("global_load_async_to_lds_b128 %0, %1, off"
                 :: "v"(loff), "v"(gptr)
                 : "memory");
}
__device__ __forceinline__ void async_wait0() {
    asm volatile("s_wait_asynccnt 0x0" ::: "memory");
}

__device__ __forceinline__ float blockReduceSum(float v, float* sh) {
    int tid = threadIdx.x;
    sh[tid] = v;
    __syncthreads();
    for (int s = 128; s > 0; s >>= 1) {
        if (tid < s) sh[tid] += sh[tid + s];
        __syncthreads();
    }
    float r = sh[0];
    __syncthreads();
    return r;
}

// ---------------- generic bf16 WMMA GEMM ----------------
// Block tile 128(M) x 64(N), K stepped by 32, double-buffered LDS.
// 8 wave32 waves in 4x2 grid; each wave computes a 32x32 strip
// = 4x v_wmma_f32_16x16x32_bf16 per K-step.
// A tile (and BT B tile) staged with global_load_async_to_lds_b128 (ASYNCcnt);
// std-B tile staged via registers + LDS transpose ([n][k] layout).
// BT=false: B row-major KxN (N dim of B buffer must be multiple of 64 / padded).
// BT=true : B row-major NxK (C = A*B^T), rows guarded by Nv.
// MODE: 0 f32(+bias) | 1 bf16(+bias) | 2 f32 exp(acc*scale) | 3 bf16 gelu(acc+bias)
//       | 4 f32 acc+bias+residual.
// Batched via grid.z: z -> (zb=z/batchH, zh=z%batchH), offsets zb*s?b + zh*s?h.

#define LSA 40   // u16 stride of LDS A rows (32 data + 8 pad) -> 80B, 16B-aligned subtiles
#define LSB 40

template<int MODE, bool BT>
__global__ __launch_bounds__(256)
void gemm_wmma(const u16* __restrict__ A, int lda, long long sAb, long long sAh,
               const u16* __restrict__ Bm, int ldb, long long sBb, long long sBh,
               void* __restrict__ Cv, int ldc, long long sCb, long long sCh,
               const float* __restrict__ bias,
               const float* __restrict__ res, int ldres, long long sRb, long long sRh,
               int K, int Mv, int Nv, int KvB, float scale, int batchH)
{
    __shared__ __align__(16) u16 lsA[2][128 * LSA];
    __shared__ __align__(16) u16 lsB[2][64 * LSB];

    int z  = blockIdx.z;
    int zb = z / batchH, zh = z % batchH;
    const u16* Ab = A  + zb * sAb + zh * sAh;
    const u16* Bb = Bm + zb * sBb + zh * sBh;
    float* Cf = (float*)Cv;
    u16*   Cu = (u16*)Cv;
    long long cOff = zb * sCb + zh * sCh;
    long long rOff = zb * sRb + zh * sRh;

    int tid  = threadIdx.x;
    int lane = tid & 31, wave = tid >> 5;
    int wm = wave >> 1, wn = wave & 1;          // 4 x 2 wave grid
    int m0blk = blockIdx.y * 128;
    int n0blk = blockIdx.x * 64;
    int lrow  = lane & 15, lhalf = lane >> 4;

    // staging thread mapping
    int arow = tid >> 2;            // 0..63   (A: 2 passes of 64 rows)
    int acs  = (tid & 3) * 8;       // 0,8,16,24
    int btn  = tid >> 2;            // BT: n row 0..63
    int btks = (tid & 3) * 8;
    int bsk  = tid >> 3;            // std B: k 0..31
    int bsn  = (tid & 7) * 8;       // std B: n 0,8,..,56

    v8f acc[4];
    #pragma unroll
    for (int i = 0; i < 4; ++i)
        acc[i] = (v8f){0.f,0.f,0.f,0.f,0.f,0.f,0.f,0.f};

    // ---- pre-zero out-of-range rows in BOTH buffers (k-invariant predicate:
    //      async copies below never touch these rows, so zeros persist) ----
    #pragma unroll
    for (int p = 0; p < 2; ++p) {
        int grow = m0blk + p * 64 + arow;
        if (grow >= Mv) {
            uint4 zz = zero4();
            *(uint4*)(&lsA[0][(p * 64 + arow) * LSA + acs]) = zz;
            *(uint4*)(&lsA[1][(p * 64 + arow) * LSA + acs]) = zz;
        }
    }
    if (BT) {
        int gn = n0blk + btn;
        if (gn >= Nv) {
            uint4 zz = zero4();
            *(uint4*)(&lsB[0][btn * LSB + btks]) = zz;
            *(uint4*)(&lsB[1][btn * LSB + btks]) = zz;
        }
    }

    // ---- prologue: issue tile k0=0 into buffer 0 ----
    uint4 rB = zero4();
    {
        #pragma unroll
        for (int p = 0; p < 2; ++p) {
            int grow = m0blk + p * 64 + arow;
            if (grow < Mv)
                async_copy_b128(&lsA[0][(p * 64 + arow) * LSA + acs],
                                Ab + (long long)grow * lda + acs);
        }
        if (BT) {
            int gn = n0blk + btn;
            if (gn < Nv)
                async_copy_b128(&lsB[0][btn * LSB + btks],
                                Bb + (long long)gn * ldb + btks);
        } else {
            if (bsk < KvB)
                rB = *(const uint4*)(Bb + (long long)bsk * ldb + n0blk + bsn);
        }
    }

    int cur = 0;
    for (int k0 = 0; k0 < K; k0 += 32) {
        // commit std-B registers into lsB[cur] (transpose to [n][k])
        if (!BT) {
            union { uint4 q; u16 h[8]; } d; d.q = rB;
            #pragma unroll
            for (int j = 0; j < 8; ++j)
                lsB[cur][(bsn + j) * LSB + bsk] = d.h[j];
        }
        async_wait0();          // tile in buf[cur] (issued last iteration) landed
        __syncthreads();        // compiler adds s_wait_dscnt before the signal

        // ---- issue tile k0+32 into buf[cur^1]; overlaps with WMMA below ----
        bool more = (k0 + 32 < K);
        if (more) {
            int k1 = k0 + 32;
            int nb = cur ^ 1;
            #pragma unroll
            for (int p = 0; p < 2; ++p) {
                int grow = m0blk + p * 64 + arow;
                if (grow < Mv)
                    async_copy_b128(&lsA[nb][(p * 64 + arow) * LSA + acs],
                                    Ab + (long long)grow * lda + k1 + acs);
            }
            if (BT) {
                int gn = n0blk + btn;
                if (gn < Nv)
                    async_copy_b128(&lsB[nb][btn * LSB + btks],
                                    Bb + (long long)gn * ldb + k1 + btks);
            } else {
                rB = zero4();
                if (bsk < KvB)
                    rB = *(const uint4*)(Bb + (long long)(k1 + bsk) * ldb + n0blk + bsn);
                if (k0 + 64 < K)   // L2 warm-up for the tile after next
                    __builtin_prefetch((const void*)(Bb + (long long)(k0 + 64 + bsk) * ldb + n0blk + bsn), 0, 1);
            }
        }

        // ---- fragments ----
        // A 16x32 bf16 (ISA table): lane = lhalf*16+lrow holds row lrow,
        //   elems 0..7 -> K = lhalf*8 + 0..7 ; elems 8..15 -> K = 16 + lhalf*8 + 0..7.
        // B 32x16 bf16 (ISA table, same as A): lane col N = lrow, elems 0..7 -> K = lhalf*8 + 0..7 ; elems 8..15 -> K = 16 + lhalf*8 + 0..7.
        union Frag { uint4 q[2]; v16bf v; } fa0, fa1, fb0, fb1;
        {
            const u16* ap0 = &lsA[cur][(wm * 32 + lrow) * LSA + lhalf * 8];
            const u16* ap1 = ap0 + 16 * LSA;
            fa0.q[0] = *(const uint4*)(ap0);
            fa0.q[1] = *(const uint4*)(ap0 + 16);
            fa1.q[0] = *(const uint4*)(ap1);
            fa1.q[1] = *(const uint4*)(ap1 + 16);
            const u16* bp0 = &lsB[cur][(wn * 32 + lrow) * LSB + lhalf * 8];
            const u16* bp1 = bp0 + 16 * LSB;
            fb0.q[0] = *(const uint4*)(bp0);
            fb0.q[1] = *(const uint4*)(bp0 + 16);
            fb1.q[0] = *(const uint4*)(bp1);
            fb1.q[1] = *(const uint4*)(bp1 + 16);
        }

        acc[0] = __builtin_amdgcn_wmma_f32_16x16x32_bf16(false, fa0.v, false, fb0.v,
                                                         (short)0, acc[0], false, false);
        acc[1] = __builtin_amdgcn_wmma_f32_16x16x32_bf16(false, fa0.v, false, fb1.v,
                                                         (short)0, acc[1], false, false);
        acc[2] = __builtin_amdgcn_wmma_f32_16x16x32_bf16(false, fa1.v, false, fb0.v,
                                                         (short)0, acc[2], false, false);
        acc[3] = __builtin_amdgcn_wmma_f32_16x16x32_bf16(false, fa1.v, false, fb1.v,
                                                         (short)0, acc[3], false, false);
        cur ^= 1;
    }

    // ---- epilogue: C/D layout: elem r -> M = lhalf*8 + r, N = lrow ----
    #pragma unroll
    for (int mt = 0; mt < 2; ++mt) {
        #pragma unroll
        for (int nt = 0; nt < 2; ++nt) {
            int col = n0blk + wn * 32 + nt * 16 + lrow;
            if (col >= Nv) continue;
            const v8f a = acc[mt * 2 + nt];
            #pragma unroll
            for (int r = 0; r < 8; ++r) {
                int row = m0blk + wm * 32 + mt * 16 + lhalf * 8 + r;
                if (row >= Mv) continue;
                float v = a[r];
                long long cidx = cOff + (long long)row * ldc + col;
                if (MODE == 0) {
                    if (bias) v += bias[col];
                    Cf[cidx] = v;
                } else if (MODE == 1) {
                    if (bias) v += bias[col];
                    Cu[cidx] = f2bf(v);
                } else if (MODE == 2) {
                    Cf[cidx] = __expf(v * scale);
                } else if (MODE == 3) {
                    v += bias[col];
                    v = 0.5f * v * (1.0f + erff(v * 0.70710678118654752f));
                    Cu[cidx] = f2bf(v);
                } else {
                    v += bias[col] + res[rOff + (long long)row * ldres + col];
                    Cf[cidx] = v;
                }
            }
        }
    }
}

// ---------------- LayerNorm over D=768, fp32 in -> bf16 out ----------------
__global__ __launch_bounds__(256)
void ln_rows(const float* __restrict__ in, long long ld,
             const float* __restrict__ g, const float* __restrict__ b,
             u16* __restrict__ out, long long ldo)
{
    __shared__ float sh[256];
    long long row = blockIdx.x;
    const float* p = in + row * ld;
    int i = threadIdx.x;
    float v0 = p[i], v1 = p[i + 256], v2 = p[i + 512];
    float s = blockReduceSum(v0 + v1 + v2, sh);
    float mean = s * (1.0f / 768.0f);
    float d0 = v0 - mean, d1 = v1 - mean, d2 = v2 - mean;
    float s2 = blockReduceSum(d0 * d0 + d1 * d1 + d2 * d2, sh);
    float inv = rsqrtf(s2 * (1.0f / 768.0f) + 1e-5f);
    u16* o = out + row * ldo;
    o[i]       = f2bf(d0 * inv * g[i]       + b[i]);
    o[i + 256] = f2bf(d1 * inv * g[i + 256] + b[i + 256]);
    o[i + 512] = f2bf(d2 * inv * g[i + 512] + b[i + 512]);
}

// ---------------- patch gather + LN1 -> bf16 A (12544 x 768) ----------------
__global__ __launch_bounds__(256)
void patch_embed_ln(const float* __restrict__ img,
                    const float* __restrict__ g, const float* __restrict__ b,
                    u16* __restrict__ out)
{
    __shared__ float sh[256];
    int row = blockIdx.x;                 // b*196 + p
    int bb = row / 196, p = row % 196;
    int ph = p / 14, pw = p % 14;
    float v[3];
    #pragma unroll
    for (int j = 0; j < 3; ++j) {
        int e  = threadIdx.x + j * 256;   // (p1*16 + p2)*3 + c
        int c  = e % 3;
        int p2 = (e / 3) % 16;
        int p1 = e / 48;
        v[j] = img[(((long long)bb * 3 + c) * 224 + ph * 16 + p1) * 224 + pw * 16 + p2];
    }
    float s = blockReduceSum(v[0] + v[1] + v[2], sh);
    float mean = s * (1.0f / 768.0f);
    float d0 = v[0] - mean, d1 = v[1] - mean, d2 = v[2] - mean;
    float s2 = blockReduceSum(d0 * d0 + d1 * d1 + d2 * d2, sh);
    float inv = rsqrtf(s2 * (1.0f / 768.0f) + 1e-5f);
    u16* o = out + (long long)row * 768;
    int i = threadIdx.x;
    o[i]       = f2bf(d0 * inv * g[i]       + b[i]);
    o[i + 256] = f2bf(d1 * inv * g[i + 256] + b[i + 256]);
    o[i + 512] = f2bf(d2 * inv * g[i + 512] + b[i + 512]);
}

// ---------------- LN2 on patch-embed + cls token + pos emb -> x fp32 ----------------
__global__ __launch_bounds__(256)
void assemble_x(const float* __restrict__ tmp, const float* __restrict__ g,
                const float* __restrict__ b, const float* __restrict__ cls,
                const float* __restrict__ pos, float* __restrict__ x)
{
    __shared__ float sh[256];
    int row = blockIdx.x;                 // b*197 + t
    int bb = row / 197, t = row % 197;
    float* xo = x + (long long)row * 768;
    int i = threadIdx.x;
    if (t == 0) {
        #pragma unroll
        for (int j = 0; j < 3; ++j) {
            int e = i + j * 256;
            xo[e] = cls[e] + pos[e];
        }
        return;
    }
    const float* p = tmp + ((long long)bb * 196 + (t - 1)) * 768;
    float v0 = p[i], v1 = p[i + 256], v2 = p[i + 512];
    float s = blockReduceSum(v0 + v1 + v2, sh);
    float mean = s * (1.0f / 768.0f);
    float d0 = v0 - mean, d1 = v1 - mean, d2 = v2 - mean;
    float s2 = blockReduceSum(d0 * d0 + d1 * d1 + d2 * d2, sh);
    float inv = rsqrtf(s2 * (1.0f / 768.0f) + 1e-5f);
    const float* pp = pos + (long long)t * 768;
    xo[i]       = d0 * inv * g[i]       + b[i]       + pp[i];
    xo[i + 256] = d1 * inv * g[i + 256] + b[i + 256] + pp[i + 256];
    xo[i + 512] = d2 * inv * g[i + 512] + b[i + 512] + pp[i + 512];
}

// ---------------- softmax normalize: numer / (sum + 1e-8) -> bf16 (K-padded to 224) ----------------
__global__ __launch_bounds__(256)
void attn_norm(const float* __restrict__ s, u16* __restrict__ a)
{
    __shared__ float sh[256];
    long long base = (long long)blockIdx.x * 224;   // blockIdx.x = bh*197 + t
    int i = threadIdx.x;
    float v = 0.f;
    if (i < 197) v = s[base + i];
    float sum = blockReduceSum(v, sh);
    float r = 1.0f / (sum + 1e-8f);
    if (i < 224) a[base + i] = f2bf(v * r);         // cols 197..223 -> 0 (K padding)
}

// ---------------- fp32 -> bf16 weight casts ----------------
__global__ __launch_bounds__(256)
void f32_to_bf16(const float* __restrict__ in, u16* __restrict__ out, long long n)
{
    long long i = (long long)blockIdx.x * 256 + threadIdx.x;
    if (i < n) out[i] = f2bf(in[i]);
}

// row-padded cast: out is rows x cout, in is rows x cin, tail columns zeroed
__global__ __launch_bounds__(256)
void f32_to_bf16_pad(const float* __restrict__ in, u16* __restrict__ out,
                     int cin, int cout, long long total)
{
    long long i = (long long)blockIdx.x * 256 + threadIdx.x;
    if (i >= total) return;
    long long r = i / cout;
    int c = (int)(i - r * cout);
    out[i] = (c < cin) ? f2bf(in[r * cin + c]) : (u16)0;
}

// ---------------- host ----------------
extern "C" void kernel_launch(void* const* d_in, const int* in_sizes, int n_in,
                              void* d_out, int out_size, void* d_ws, size_t ws_size,
                              hipStream_t stream)
{
    (void)in_sizes; (void)n_in; (void)out_size; (void)ws_size;

    const float* img    = (const float*)d_in[0];
    const float* ln1g   = (const float*)d_in[1];
    const float* ln1b   = (const float*)d_in[2];
    const float* patchw = (const float*)d_in[3];
    const float* patchb = (const float*)d_in[4];
    const float* ln2g   = (const float*)d_in[5];
    const float* ln2b   = (const float*)d_in[6];
    const float* pos    = (const float*)d_in[7];
    const float* cls    = (const float*)d_in[8];
    const float* alng   = (const float*)d_in[9];
    const float* alnb   = (const float*)d_in[10];
    const float* wqkv   = (const float*)d_in[11];
    const float* wout   = (const float*)d_in[12];
    const float* bout   = (const float*)d_in[13];
    const float* flng   = (const float*)d_in[14];
    const float* flnb   = (const float*)d_in[15];
    const float* fw1    = (const float*)d_in[16];
    const float* fb1    = (const float*)d_in[17];
    const float* fw2    = (const float*)d_in[18];
    const float* fb2    = (const float*)d_in[19];
    const float* fing   = (const float*)d_in[20];
    const float* finb   = (const float*)d_in[21];
    const float* headw  = (const float*)d_in[22];
    const float* headb  = (const float*)d_in[23];

    const long long MT = 64LL * 197;   // 12608 token rows
    const long long MP = 64LL * 196;   // 12544 patch rows
    const int NT = 197, SK = 224;

    char* ws = (char*)d_ws;
    size_t off = 0;
    auto alloc = [&](size_t bytes) {
        size_t o = off;
        off += (bytes + 255) & ~(size_t)255;
        return o;
    };
    float* x    = (float*)(ws + alloc(MT * 768 * 4));
    u16*   xn   = (u16*)  (ws + alloc(MT * 768 * 2));
    u16*   qkv  = (u16*)  (ws + alloc(MT * 2304 * 2));
    float* sc   = (float*)(ws + alloc(768LL * NT * SK * 4));  // scores; also patch-GEMM fp32 tmp
    u16*   attn = (u16*)  (ws + alloc(768LL * NT * SK * 2));
    u16*   hid  = (u16*)  (ws + alloc(MT * 3072 * 2));
    u16*   ao   = (u16*)  (ws + alloc(MT * 768 * 2));
    u16*   pwb  = (u16*)  (ws + alloc(768LL * 768 * 2));
    u16*   wqkvb= (u16*)  (ws + alloc(12LL * 768 * 2304 * 2));
    u16*   woutb= (u16*)  (ws + alloc(12LL * 768 * 768 * 2));
    u16*   fw1b = (u16*)  (ws + alloc(12LL * 768 * 3072 * 2));
    u16*   fw2b = (u16*)  (ws + alloc(12LL * 3072 * 768 * 2));
    u16*   hwb  = (u16*)  (ws + alloc(768LL * 1024 * 2));     // head_w padded 1000 -> 1024

    dim3 blk(256);
    auto cv = [&](const float* s_, u16* d_, long long n) {
        f32_to_bf16<<<dim3((unsigned)((n + 255) / 256)), blk, 0, stream>>>(s_, d_, n);
    };
    cv(patchw, pwb,   768LL * 768);
    cv(wqkv,   wqkvb, 12LL * 768 * 2304);
    cv(wout,   woutb, 12LL * 768 * 768);
    cv(fw1,    fw1b,  12LL * 768 * 3072);
    cv(fw2,    fw2b,  12LL * 3072 * 768);
    {
        long long tot = 768LL * 1024;
        f32_to_bf16_pad<<<dim3((unsigned)((tot + 255) / 256)), blk, 0, stream>>>(
            headw, hwb, 1000, 1024, tot);
    }

    // patch embed: gather+LN1 -> xn, GEMM (12544x768x768) -> sc (fp32 tmp), LN2+cls+pos -> x
    patch_embed_ln<<<dim3(12544), blk, 0, stream>>>(img, ln1g, ln1b, xn);
    gemm_wmma<0, false><<<dim3(12, 98, 1), blk, 0, stream>>>(
        xn, 768, 0, 0, pwb, 768, 0, 0, (void*)sc, 768, 0, 0,
        patchb, nullptr, 0, 0, 0, 768, (int)MP, 768, 768, 1.0f, 1);
    assemble_x<<<dim3((unsigned)MT), blk, 0, stream>>>(sc, ln2g, ln2b, cls, pos, x);

    for (int l = 0; l < 12; ++l) {
        // attention
        ln_rows<<<dim3((unsigned)MT), blk, 0, stream>>>(x, 768, alng + l * 768, alnb + l * 768, xn, 768);
        gemm_wmma<1, false><<<dim3(36, 99, 1), blk, 0, stream>>>(
            xn, 768, 0, 0, wqkvb + (long long)l * 768 * 2304, 2304, 0, 0,
            (void*)qkv, 2304, 0, 0, nullptr, nullptr, 0, 0, 0,
            768, (int)MT, 2304, 768, 1.0f, 1);
        // scores = exp(q @ k^T * 1/8), batched over (b,h)
        gemm_wmma<2, true><<<dim3(4, 2, 768), blk, 0, stream>>>(
            qkv, 2304, (long long)NT * 2304, 64,
            qkv + 768, 2304, (long long)NT * 2304, 64,
            (void*)sc, SK, 12LL * NT * SK, (long long)NT * SK,
            nullptr, nullptr, 0, 0, 0,
            64, NT, NT, 64, 0.125f, 12);
        attn_norm<<<dim3(768 * NT), blk, 0, stream>>>(sc, attn);
        // context = attn @ v  -> ao[b, t, h*64 + d]
        gemm_wmma<1, false><<<dim3(1, 2, 768), blk, 0, stream>>>(
            attn, SK, 12LL * NT * SK, (long long)NT * SK,
            qkv + 1536, 2304, (long long)NT * 2304, 64,
            (void*)ao, 768, (long long)NT * 768, 64,
            nullptr, nullptr, 0, 0, 0,
            SK, NT, 64, NT, 1.0f, 12);
        // out proj + bias + residual
        gemm_wmma<4, false><<<dim3(12, 99, 1), blk, 0, stream>>>(
            ao, 768, 0, 0, woutb + (long long)l * 768 * 768, 768, 0, 0,
            (void*)x, 768, 0, 0, bout + l * 768, x, 768, 0, 0,
            768, (int)MT, 768, 768, 1.0f, 1);
        // MLP
        ln_rows<<<dim3((unsigned)MT), blk, 0, stream>>>(x, 768, flng + l * 768, flnb + l * 768, xn, 768);
        gemm_wmma<3, false><<<dim3(48, 99, 1), blk, 0, stream>>>(
            xn, 768, 0, 0, fw1b + (long long)l * 768 * 3072, 3072, 0, 0,
            (void*)hid, 3072, 0, 0, fb1 + l * 3072, nullptr, 0, 0, 0,
            768, (int)MT, 3072, 768, 1.0f, 1);
        gemm_wmma<4, false><<<dim3(12, 99, 1), blk, 0, stream>>>(
            hid, 3072, 0, 0, fw2b + (long long)l * 3072 * 768, 768, 0, 0,
            (void*)x, 768, 0, 0, fb2 + l * 768, x, 768, 0, 0,
            3072, (int)MT, 768, 3072, 1.0f, 1);
    }

    // final LN on cls tokens (row b*197) -> 64x768 bf16, then head GEMM 64x768x1000 -> d_out fp32
    ln_rows<<<dim3(64), blk, 0, stream>>>(x, 197LL * 768, fing, finb, xn, 768);
    gemm_wmma<0, false><<<dim3(16, 1, 1), blk, 0, stream>>>(
        xn, 768, 0, 0, hwb, 1024, 0, 0,
        d_out, 1000, 0, 0, headb, nullptr, 0, 0, 0,
        768, 64, 1000, 768, 1.0f, 1);
}
